// DeepseekV3Attention_88630945120713
// MI455X (gfx1250) — matrix operand order repaired and checked by reading, not compiled, not run on hardware
//
#include <hip/hip_runtime.h>
#include <hip/hip_bf16.h>

// ---------------- problem constants ----------------
constexpr int HID  = 2048;
constexpr int NH   = 16;
constexpr int QLR  = 1536;
constexpr int KVLR = 512;
constexpr int DN   = 128;
constexpr int DR   = 64;
constexpr int DQK  = DN + DR;   // 192
constexpr int DV   = 128;
constexpr int Bc   = 2;
constexpr int Tc   = 2048;
constexpr size_t ROWS = (size_t)Bc * Tc;          // 4096 tokens
constexpr int NQK = NH * DQK;                     // 3072
constexpr int NKV = NH * (DN + DV);               // 4096
constexpr int NDV = NH * DV;                      // 2048

typedef __bf16 bf16_t;
typedef bf16_t v16bf __attribute__((ext_vector_type(16)));
typedef bf16_t v8bf  __attribute__((ext_vector_type(8)));
typedef float  v8f   __attribute__((ext_vector_type(8)));

// ---------------- WMMA fragment helpers (wave32, 16x16x32 bf16) ----------------
// A (16x32, row-major): lane m = lane&15; lanes<16 take K 0..15,
// lanes>=16 take K 16..31.
__device__ inline v16bf load_a_frag(const bf16_t* base, int ld) {
  const int lane = threadIdx.x & 31;
  const int row  = lane & 15;
  const int k0   = (lane >> 4) << 4;     // 0 or 16
  const bf16_t* p = base + (size_t)row * ld;
  union { v16bf v; v8bf h[2]; } u;
  u.h[0] = *reinterpret_cast<const v8bf*>(p + k0);
  u.h[1] = *reinterpret_cast<const v8bf*>(p + k0 + 8);
  return u.v;
}

// B (32x16) from B^T stored row-major [N, K]: lane n = lane&15 holds K 0..15
// (lanes<16) or K 16..31 (lanes>=16) of column n -> contiguous 32B chunk.
__device__ inline v16bf load_b_frag(const bf16_t* base, int ld) {
  const int lane = threadIdx.x & 31;
  const int col  = lane & 15;
  const int k0   = (lane >> 4) << 4;     // 0 or 16
  const bf16_t* p = base + (size_t)col * ld + k0;
  union { v16bf v; v8bf h[2]; } u;
  u.h[0] = *reinterpret_cast<const v8bf*>(p);
  u.h[1] = *reinterpret_cast<const v8bf*>(p + 8);
  return u.v;
}

__device__ inline v8f wmma_bf16(v16bf a, v16bf b, v8f c) {
  return __builtin_amdgcn_wmma_f32_16x16x32_bf16(false, a, false, b, (short)0, c,
                                                 false, false);
}

// ---------------- CDNA5 async global->LDS copy (ASYNCcnt path) ----------------
__device__ inline void async_copy_b128(bf16_t* lds_dst, const bf16_t* gsrc) {
  const unsigned lds_off = (unsigned)(unsigned long long)lds_dst;  // low 32b = LDS addr
  asm volatile("global_load_async_to_lds_b128 %0, %1, off"
               :: "v"(lds_off), "v"(gsrc) : "memory");
}
__device__ inline void async_wait0() {
  asm volatile("s_wait_asynccnt 0x0" ::: "memory");
}

// ---------------- elementwise / layout kernels ----------------
__global__ __launch_bounds__(256) void k_f32_to_bf16(const float* __restrict__ in,
                                                     bf16_t* __restrict__ out, size_t n) {
  size_t i = (size_t)blockIdx.x * blockDim.x + threadIdx.x;
  if (i < n) out[i] = (bf16_t)in[i];
}

// in: f32 [K, N] row-major  ->  out: bf16 [N, K] row-major
__global__ __launch_bounds__(256) void k_transpose_bf16(const float* __restrict__ in,
                                                        bf16_t* __restrict__ out,
                                                        int K, int N) {
  size_t i = (size_t)blockIdx.x * blockDim.x + threadIdx.x;
  size_t total = (size_t)K * N;
  if (i < total) {
    int kk = (int)(i / N);
    int nn = (int)(i % N);
    out[(size_t)nn * K + kk] = (bf16_t)in[i];
  }
}

// rmsnorm of first C elems of each row; in f32 stride sin, out bf16 stride sout
__global__ __launch_bounds__(256) void k_rmsnorm(const float* __restrict__ in, int sin,
                                                 const float* __restrict__ scale,
                                                 bf16_t* __restrict__ out, int sout, int C) {
  __shared__ float red[256];
  const int row = blockIdx.x;
  const float* p = in + (size_t)row * sin;
  float s = 0.f;
  for (int i = threadIdx.x; i < C; i += 256) { float v = p[i]; s += v * v; }
  red[threadIdx.x] = s;
  __syncthreads();
  for (int off = 128; off > 0; off >>= 1) {
    if ((int)threadIdx.x < off) red[threadIdx.x] += red[threadIdx.x + off];
    __syncthreads();
  }
  const float r = rsqrtf(red[0] / (float)C + 1e-6f);
  bf16_t* q = out + (size_t)row * sout;
  for (int i = threadIdx.x; i < C; i += 256) q[i] = (bf16_t)(p[i] * r * scale[i]);
}

__device__ inline void rope_pair(float fpos, int f, float& c, float& sn) {
  // inv_freq = theta^(-2f/64) = exp(-f/32 * ln(10000))
  const float ang = fpos * __expf((float)f * (-9.210340371976184f / 32.0f));
  c = __cosf(ang); sn = __sinf(ang);
}

// qfull f32 [ROWS, NH*DQK] -> q bf16 [B][NH][T][DQK] with RoPE on last DR dims
__global__ __launch_bounds__(256) void k_build_q(const float* __restrict__ qfull,
                                                 const int* __restrict__ pos,
                                                 bf16_t* __restrict__ qout) {
  const int row = blockIdx.x;                 // b*T + t
  const int b = row / Tc, t = row % Tc;
  const float fpos = (float)pos[row];
  for (int i = threadIdx.x; i < NQK; i += 256) {
    const int h = i / DQK, d = i % DQK;
    const float* src = qfull + (size_t)row * NQK + (size_t)h * DQK;
    float v;
    if (d < DN) {
      v = src[d];
    } else {
      const int j = d - DN;
      const int f = (j < 32) ? j : j - 32;
      float c, sn; rope_pair(fpos, f, c, sn);
      const float x1 = src[DN + f], x2 = src[DN + 32 + f];
      v = (j < 32) ? (x1 * c - x2 * sn) : (x2 * c + x1 * sn);
    }
    qout[(((size_t)(b * NH + h)) * Tc + t) * DQK + d] = (bf16_t)v;
  }
}

// kv f32 [ROWS, NH*(DN+DV)], ckv f32 [ROWS, KVLR+DR]  ->
//   k bf16 [B][NH][T][DQK] (k_pass | roped k_rot broadcast), v^T bf16 [B][NH][DV][T]
__global__ __launch_bounds__(256) void k_build_kv(const float* __restrict__ kv,
                                                  const float* __restrict__ ckv,
                                                  const int* __restrict__ pos,
                                                  bf16_t* __restrict__ kout,
                                                  bf16_t* __restrict__ vt) {
  const int row = blockIdx.x;
  const int b = row / Tc, t = row % Tc;
  const float fpos = (float)pos[row];
  const float* kvrow  = kv  + (size_t)row * NKV;
  const float* rotsrc = ckv + (size_t)row * (KVLR + DR) + KVLR;
  for (int i = threadIdx.x; i < NQK; i += 256) {
    const int h = i / DQK, d = i % DQK;
    float v;
    if (d < DN) {
      v = kvrow[h * (DN + DV) + d];
    } else {
      const int j = d - DN;
      const int f = (j < 32) ? j : j - 32;
      float c, sn; rope_pair(fpos, f, c, sn);
      const float x1 = rotsrc[f], x2 = rotsrc[32 + f];
      v = (j < 32) ? (x1 * c - x2 * sn) : (x2 * c + x1 * sn);
    }
    kout[(((size_t)(b * NH + h)) * Tc + t) * DQK + d] = (bf16_t)v;
  }
  for (int i = threadIdx.x; i < NDV; i += 256) {
    const int h = i / DV, d = i % DV;
    vt[(((size_t)(b * NH + h)) * DV + d) * Tc + t] =
        (bf16_t)kvrow[h * (DN + DV) + DN + d];
  }
}

// ---------------- bf16 WMMA GEMM: C[M,N] f32 = A[M,K] * B^T[N,K] ----------------
// block = 256 threads (8 waves) stacked on M: block tile 128M x 64N, wave 16M x 64N.
// B tile (64x32 bf16) fetched ONCE per block via async global->LDS (double buffered,
// ASYNCcnt), shared by all 8 waves. A is per-wave, register-prefetched.
constexpr int BSTRIDE = 40;   // halves per staged B row (80B: 16B-aligned, conflict-free)

__device__ inline void stage_b(const bf16_t* __restrict__ Bt, int K, int n0, int k,
                               bf16_t* dst) {
  const int tid  = threadIdx.x;        // 0..255 -> one 16B chunk each (4KB tile)
  const int row  = tid >> 2;           // 0..63
  const int part = tid & 3;            // 4 x 8 halves per row
  async_copy_b128(dst + row * BSTRIDE + part * 8,
                  Bt + (size_t)(n0 + row) * K + k + part * 8);
}

__device__ inline v16bf load_b_lds(const bf16_t* buf, int sub, int stride) {
  const int lane = threadIdx.x & 31;
  const int col  = lane & 15;
  const int k0   = (lane >> 4) << 4;   // 0 or 16
  const bf16_t* p = buf + (sub * 16 + col) * stride + k0;
  union { v16bf v; v8bf h[2]; } u;
  u.h[0] = *reinterpret_cast<const v8bf*>(p);
  u.h[1] = *reinterpret_cast<const v8bf*>(p + 8);
  return u.v;
}

__global__ __launch_bounds__(256) void k_gemm_bf16(const bf16_t* __restrict__ A,
                                                   const bf16_t* __restrict__ Bt,
                                                   float* __restrict__ C,
                                                   int M, int N, int K) {
  __shared__ __align__(16) bf16_t Bs[2][64 * BSTRIDE];
  const int wid = threadIdx.x >> 5;
  const int m0 = blockIdx.y * 128 + wid * 16;
  const int n0 = blockIdx.x * 64;

  v8f acc[4] = {v8f{}, v8f{}, v8f{}, v8f{}};

  stage_b(Bt, K, n0, 0, Bs[0]);
  v16bf a = load_a_frag(A + (size_t)m0 * K, K);
  async_wait0();
  __syncthreads();

  int cur = 0;
  for (int k = 0; k < K; k += 32) {
    v16bf an = a;
    if (k + 32 < K) {
      stage_b(Bt, K, n0, k + 32, Bs[cur ^ 1]);          // overlap with WMMAs below
      an = load_a_frag(A + (size_t)m0 * K + k + 32, K); // register prefetch
    }
#pragma unroll
    for (int j = 0; j < 4; ++j)
      acc[j] = wmma_bf16(a, load_b_lds(Bs[cur], j, BSTRIDE), acc[j]);
    async_wait0();      // our slice of next tile landed in LDS
    __syncthreads();    // everyone's slice landed; cur buffer free to overwrite
    a = an;
    cur ^= 1;
  }

  const int lane = threadIdx.x & 31;
  const int n  = lane & 15;
  const int mb = (lane >> 4) << 3;
#pragma unroll
  for (int j = 0; j < 4; ++j)
#pragma unroll
    for (int r = 0; r < 8; ++r)
      C[(size_t)(m0 + mb + r) * N + n0 + j * 16 + n] = acc[j][r];
}

// ---------------- cooperative flash attention ----------------
// block = 4 waves = 64 consecutive queries of one (b,h). K/V key-blocks (32 keys)
// are async-staged to LDS once per block, double-buffered, shared by all 4 waves.
// q,k bf16 [B*NH][T][DQK]; v^T bf16 [B*NH][DV][T]; out bf16 [B*T][NH*DV]
constexpr int KSTR = 200;   // LDS stride (halves) of a staged K row (DQK=192 + pad)
constexpr int VSTR = 40;    // LDS stride (halves) of a staged V^T row (32 keys + pad)

__global__ __launch_bounds__(128) void k_attention(const bf16_t* __restrict__ Q,
                                                   const bf16_t* __restrict__ Kk,
                                                   const bf16_t* __restrict__ Vt,
                                                   bf16_t* __restrict__ Out) {
  const int qi64 = blockIdx.x;        // 64-query block
  const int h    = blockIdx.y;
  const int b    = blockIdx.z;
  const int bh   = b * NH + h;
  const int wid  = threadIdx.x >> 5;  // wave -> 16-query tile within the 64
  const int qt   = qi64 * 64 + wid * 16;
  const bf16_t* qbase = Q  + ((size_t)bh * Tc + qt) * DQK;
  const bf16_t* kbase = Kk + (size_t)bh * Tc * DQK;
  const bf16_t* vbase = Vt + (size_t)bh * DV * Tc;

  __shared__ __align__(16) bf16_t Ks[2][32 * KSTR];   // 32 keys x DQK
  __shared__ __align__(16) bf16_t Vs[2][128 * VSTR];  // DV x 32 keys
  __shared__ __align__(16) bf16_t Pl[4][16 * 32];     // per-wave P staging

  const int lane = threadIdx.x & 31;
  const int n  = lane & 15;
  const int mb = (lane >> 4) << 3;
  const float scale = 0.07216878364870322f;   // 1/sqrt(192)

  v16bf qf[6];
#pragma unroll
  for (int j = 0; j < 6; ++j) qf[j] = load_a_frag(qbase + j * 32, DQK);

  v8f o[8];
#pragma unroll
  for (int j = 0; j < 8; ++j) o[j] = v8f{};
  float mrow[8], lrow[8];
#pragma unroll
  for (int r = 0; r < 8; ++r) { mrow[r] = -3.0e38f; lrow[r] = 0.f; }

  // cooperative staging of one 32-key block (K: 768 chunks, V: 512 chunks; 128 thr)
  auto stage_kv = [&](int kb, int buf) {
    const int tid = threadIdx.x;
#pragma unroll
    for (int c = 0; c < 6; ++c) {                 // K tile: 32 x 192 halves
      const int idx  = tid + c * 128;             // 0..767
      const int row  = idx / 24;                  // key
      const int part = idx % 24;                  // 8-half chunk within row
      async_copy_b128(&Ks[buf][row * KSTR + part * 8],
                      kbase + (size_t)(kb + row) * DQK + part * 8);
    }
#pragma unroll
    for (int c = 0; c < 4; ++c) {                 // V tile: 128 x 32 halves
      const int idx  = tid + c * 128;             // 0..511
      const int row  = idx / 4;                   // dim
      const int part = idx % 4;
      async_copy_b128(&Vs[buf][row * VSTR + part * 8],
                      vbase + (size_t)row * Tc + kb + part * 8);
    }
  };

  const int qend = qt + 15;
  const int nblocks = 2 * (qi64 + 1);             // key blocks for this 64-q block
  stage_kv(0, 0);
  async_wait0();
  __syncthreads();

  int cur = 0;
  for (int blk = 0; blk < nblocks; ++blk) {
    const int kb0 = blk * 32;
    if (blk + 1 < nblocks) stage_kv(kb0 + 32, cur ^ 1);  // overlap with compute

    if (kb0 <= qend) {                  // wave-uniform causal guard
      v8f s0 = v8f{}, s1 = v8f{};
#pragma unroll
      for (int j = 0; j < 6; ++j)
        s0 = wmma_bf16(qf[j], load_b_lds(&Ks[cur][j * 32], 0, KSTR), s0);
      const bool have1 = (kb0 + 16) <= qend;
      if (have1) {
#pragma unroll
        for (int j = 0; j < 6; ++j)
          s1 = wmma_bf16(qf[j], load_b_lds(&Ks[cur][j * 32], 1, KSTR), s1);
      }
      // online softmax (rows live across the 16-lane groups; 8 rows per lane)
#pragma unroll
      for (int r = 0; r < 8; ++r) {
        const int qrow = qt + mb + r;
        float v0 = s0[r] * scale;
        if (kb0 + n > qrow) v0 = -3.0e38f;
        float v1 = have1 ? s1[r] * scale : -3.0e38f;
        if (kb0 + 16 + n > qrow) v1 = -3.0e38f;
        float mx = fmaxf(v0, v1);
        for (int off = 1; off < 16; off <<= 1)
          mx = fmaxf(mx, __shfl_xor(mx, off, 32));
        const float mnew = fmaxf(mrow[r], mx);
        const float corr = __expf(mrow[r] - mnew);
        mrow[r] = mnew;
        const float p0 = __expf(v0 - mnew);
        const float p1 = __expf(v1 - mnew);
        float rs = p0 + p1;
        for (int off = 1; off < 16; off <<= 1)
          rs += __shfl_xor(rs, off, 32);
        lrow[r] = lrow[r] * corr + rs;
#pragma unroll
        for (int j = 0; j < 8; ++j) o[j][r] *= corr;
        Pl[wid][(mb + r) * 32 + n]      = (bf16_t)p0;
        Pl[wid][(mb + r) * 32 + 16 + n] = (bf16_t)p1;
      }
      asm volatile("s_wait_dscnt 0x0" ::: "memory");   // per-wave P region ready
      const v16bf pa = load_a_frag(&Pl[wid][0], 32);
#pragma unroll
      for (int j = 0; j < 8; ++j)
        o[j] = wmma_bf16(pa, load_b_lds(&Vs[cur][0], j, VSTR), o[j]);
    }

    async_wait0();      // our slice of next K/V tile landed
    __syncthreads();    // all slices landed; cur buffer reusable
    cur ^= 1;
  }

#pragma unroll
  for (int r = 0; r < 8; ++r) {
    const float inv = 1.0f / lrow[r];
    const int trow = qt + mb + r;
    const size_t orow = ((size_t)b * Tc + trow) * NDV + (size_t)h * DV;
#pragma unroll
    for (int j = 0; j < 8; ++j)
      Out[orow + j * 16 + n] = (bf16_t)(o[j][r] * inv);
  }
}

// ---------------- host orchestration ----------------
static inline size_t alloc_off(size_t& o, size_t bytes) {
  size_t r = o; o += (bytes + 255) & ~(size_t)255; return r;
}

extern "C" void kernel_launch(void* const* d_in, const int* in_sizes, int n_in,
                              void* d_out, int out_size, void* d_ws, size_t ws_size,
                              hipStream_t stream) {
  (void)in_sizes; (void)n_in; (void)out_size; (void)ws_size;
  const float* x         = (const float*)d_in[0];
  const int*   positions = (const int*)d_in[2];
  const float* wqa       = (const float*)d_in[3];
  const float* qa_scale  = (const float*)d_in[4];
  const float* wqb       = (const float*)d_in[5];
  const float* wkva      = (const float*)d_in[6];
  const float* kva_scale = (const float*)d_in[7];
  const float* wkvb      = (const float*)d_in[8];
  const float* wo        = (const float*)d_in[9];
  float* out = (float*)d_out;

  char* ws = (char*)d_ws;
  size_t o = 0;
  bf16_t* xbf    = (bf16_t*)(ws + alloc_off(o, ROWS * HID * 2));
  bf16_t* wqaT   = (bf16_t*)(ws + alloc_off(o, (size_t)QLR * HID * 2));
  bf16_t* wqbT   = (bf16_t*)(ws + alloc_off(o, (size_t)NQK * QLR * 2));
  bf16_t* wkvaT  = (bf16_t*)(ws + alloc_off(o, (size_t)(KVLR + DR) * HID * 2));
  bf16_t* wkvbT  = (bf16_t*)(ws + alloc_off(o, (size_t)NKV * KVLR * 2));
  bf16_t* woT    = (bf16_t*)(ws + alloc_off(o, (size_t)HID * NDV * 2));
  bf16_t* cqn    = (bf16_t*)(ws + alloc_off(o, ROWS * QLR * 2));
  bf16_t* qattn  = (bf16_t*)(ws + alloc_off(o, ROWS * NH * DQK * 2));
  bf16_t* kvlatn = (bf16_t*)(ws + alloc_off(o, ROWS * KVLR * 2));
  bf16_t* kattn  = (bf16_t*)(ws + alloc_off(o, ROWS * NH * DQK * 2));
  bf16_t* vt     = (bf16_t*)(ws + alloc_off(o, ROWS * NH * DV * 2));
  bf16_t* attnout= (bf16_t*)(ws + alloc_off(o, ROWS * NDV * 2));
  float*  f32a   = (float*)(ws + alloc_off(o, ROWS * (size_t)NKV * 4)); // cq/qfull/kvfull
  float*  f32b   = (float*)(ws + alloc_off(o, ROWS * (size_t)(KVLR + DR) * 4)); // ckv

  const dim3 blk256(256);
  {
    size_t n = ROWS * HID;
    k_f32_to_bf16<<<dim3((unsigned)((n + 255) / 256)), blk256, 0, stream>>>(x, xbf, n);
  }
  k_transpose_bf16<<<dim3((unsigned)(((size_t)HID * QLR + 255) / 256)), blk256, 0, stream>>>(wqa, wqaT, HID, QLR);
  k_transpose_bf16<<<dim3((unsigned)(((size_t)QLR * NQK + 255) / 256)), blk256, 0, stream>>>(wqb, wqbT, QLR, NQK);
  k_transpose_bf16<<<dim3((unsigned)(((size_t)HID * (KVLR + DR) + 255) / 256)), blk256, 0, stream>>>(wkva, wkvaT, HID, KVLR + DR);
  k_transpose_bf16<<<dim3((unsigned)(((size_t)KVLR * NKV + 255) / 256)), blk256, 0, stream>>>(wkvb, wkvbT, KVLR, NKV);
  k_transpose_bf16<<<dim3((unsigned)(((size_t)NDV * HID + 255) / 256)), blk256, 0, stream>>>(wo, woT, NDV, HID);

  // q path: cq = x @ wqa ; rmsnorm ; qfull = cqn @ wqb ; rope + layout
  k_gemm_bf16<<<dim3(QLR / 64, (unsigned)(ROWS / 128)), dim3(256), 0, stream>>>(xbf, wqaT, f32a, (int)ROWS, QLR, HID);
  k_rmsnorm<<<dim3((unsigned)ROWS), blk256, 0, stream>>>(f32a, QLR, qa_scale, cqn, QLR, QLR);
  k_gemm_bf16<<<dim3(NQK / 64, (unsigned)(ROWS / 128)), dim3(256), 0, stream>>>(cqn, wqbT, f32a, (int)ROWS, NQK, QLR);
  k_build_q<<<dim3((unsigned)ROWS), blk256, 0, stream>>>(f32a, positions, qattn);

  // kv path: ckv = x @ wkva ; rmsnorm(512) ; kv = kvlatn @ wkvb ; build k/v^T
  k_gemm_bf16<<<dim3((KVLR + DR) / 64, (unsigned)(ROWS / 128)), dim3(256), 0, stream>>>(xbf, wkvaT, f32b, (int)ROWS, KVLR + DR, HID);
  k_rmsnorm<<<dim3((unsigned)ROWS), blk256, 0, stream>>>(f32b, KVLR + DR, kva_scale, kvlatn, KVLR, KVLR);
  k_gemm_bf16<<<dim3(NKV / 64, (unsigned)(ROWS / 128)), dim3(256), 0, stream>>>(kvlatn, wkvbT, f32a, (int)ROWS, NKV, KVLR);
  k_build_kv<<<dim3((unsigned)ROWS), blk256, 0, stream>>>(f32a, f32b, positions, kattn, vt);

  // cooperative flash attention: 4 waves / 64 queries per block, shared K/V via LDS
  k_attention<<<dim3(Tc / 64, NH, Bc), dim3(128), 0, stream>>>(qattn, kattn, vt, attnout);

  // output projection
  k_gemm_bf16<<<dim3(HID / 64, (unsigned)(ROWS / 128)), dim3(256), 0, stream>>>(attnout, woT, out, (int)ROWS, HID, NDV);
}
